// HybridLayer_50242527428794
// MI455X (gfx1250) — matrix operand order repaired and checked by reading, not compiled, not run on hardware
//
#include <hip/hip_runtime.h>
#include <hip/hip_bf16.h>
#include <stdint.h>

// ---------------------------------------------------------------------------
// Problem constants (from reference): x[B=64,C=256,H=32,W=32], N=B*H*W=65536
// nodes, 256 channels, 3x3 conv, GraphConv with E edges (E read from in_sizes).
// ---------------------------------------------------------------------------
#define Bb   64
#define Cc   256
#define Hh   32
#define Ww   32
#define Nn   (Bb*Hh*Ww)        // 65536 rows / nodes
#define HW   (Hh*Ww)           // 1024
#define KCONV_STEPS 72         // 9 taps * (256/32) k-blocks of 32
#define KGRAPH_STEPS 16        // 512 / 32

typedef __attribute__((ext_vector_type(16))) _Float16 v16h;
typedef __attribute__((ext_vector_type(8)))  float    v8f;

union FragH { v16h h; uint4 u[2]; };
union H4    { _Float16 h[4]; uint2 u; };

// ---------------------------------------------------------------------------
// Workspace layout (bytes)
// ---------------------------------------------------------------------------
static const size_t O_XH    = 0;                            // NHWC f16 x        : N*C*2   = 32 MB
static const size_t O_CRAW  = O_XH    + (size_t)Nn*Cc*2;    // conv raw f32 (reused as graph raw) : 64 MB
static const size_t O_FEAT  = O_CRAW  + (size_t)Nn*Cc*4;    // [agg_f16 | nodes_f16] : N*512*2 = 64 MB
static const size_t O_AGG   = O_FEAT  + (size_t)Nn*512*2;   // agg f32           : 64 MB
static const size_t O_WCONV = O_AGG   + (size_t)Nn*Cc*4;    // packed conv W f16 : 9*256*256*2
static const size_t O_WG    = O_WCONV + (size_t)9*Cc*Cc*2;  // packed graph W f16: 512*256*2
static const size_t O_STATS = O_WG    + (size_t)512*Cc*2;   // 2048 floats stats + 64-float zero page

// stats float offsets
#define S_SUM2D   0
#define S_SQ2D    256
#define S_SC2D    512
#define S_SH2D    768
#define S_SUM1D   1024
#define S_SQ1D    1280
#define S_SC1D    1536
#define S_SH1D    1792
#define S_ZPAGE   2048          // 64 floats, stays zero: async-load source for OOB lanes

// ---------------------------------------------------------------------------
// Utility: zero a float buffer (grid-stride)
// ---------------------------------------------------------------------------
__global__ void k_zero(float* __restrict__ p, int n) {
    for (int i = blockIdx.x * blockDim.x + threadIdx.x; i < n; i += gridDim.x * blockDim.x)
        p[i] = 0.0f;
}

// ---------------------------------------------------------------------------
// Pack conv weights OIHW f32 -> per-lane WMMA B-fragment order, f16.
// Fragment block = 32(K=cin) x 16(N=cout). Block id = (tap*8 + kb)*16 + nb.
// Within block: lane = nn + 16*(kk>=16), 16 halfs per lane, half j = kk&15.
// ---------------------------------------------------------------------------
__global__ void k_pack_conv_w(const float* __restrict__ w, _Float16* __restrict__ wp) {
    int idx = blockIdx.x * blockDim.x + threadIdx.x;
    if (idx >= 9 * Cc * Cc) return;
    int co  = idx / (Cc * 9);
    int rem = idx % (Cc * 9);
    int ci  = rem / 9;
    int t   = rem % 9;                  // tap = ky*3+kx
    float v = w[idx];                   // w[co][ci][ky][kx] row-major == idx
    int kb = ci >> 5, kk = ci & 31;
    int nb = co >> 4, nn = co & 15;
    int lane = nn + ((kk & 16) ? 16 : 0);
    int j = kk & 15;
    size_t blk = (size_t)((t * 8 + kb) * 16 + nb);
    wp[blk * 512 + lane * 16 + j] = (_Float16)v;
}

// Pack [w_rel ; w_root] (each [256,256] row-major, out x in) into K=512 blocks.
__global__ void k_pack_graph_w(const float* __restrict__ wrel, const float* __restrict__ wroot,
                               _Float16* __restrict__ wp) {
    int idx = blockIdx.x * blockDim.x + threadIdx.x;
    if (idx >= 512 * Cc) return;
    int co = idx >> 9;
    int k  = idx & 511;
    float v = (k < 256) ? wrel[co * 256 + k] : wroot[co * 256 + (k - 256)];
    int kb = k >> 5, kk = k & 31;
    int nb = co >> 4, nn = co & 15;
    int lane = nn + ((kk & 16) ? 16 : 0);
    int j = kk & 15;
    size_t blk = (size_t)(kb * 16 + nb);
    wp[blk * 512 + lane * 16 + j] = (_Float16)v;
}

// ---------------------------------------------------------------------------
// NCHW f32 -> NHWC f16 (coalesced read)
// ---------------------------------------------------------------------------
__global__ void k_x_to_nhwc(const float* __restrict__ x, _Float16* __restrict__ xh) {
    int idx = blockIdx.x * blockDim.x + threadIdx.x;   // over B*C*H*W
    if (idx >= Bb * Cc * HW) return;
    int hw = idx & (HW - 1);
    int c  = (idx >> 10) & (Cc - 1);
    int b  = idx >> 18;
    int node = (b << 10) | hw;
    xh[(size_t)node * Cc + c] = (_Float16)x[idx];
}

// ---------------------------------------------------------------------------
// gfx1250 async copy: global -> LDS (ASYNCcnt-tracked, no VGPR staging)
// ---------------------------------------------------------------------------
__device__ __forceinline__ void async_g2l_b128(const _Float16* g, _Float16* lds) {
    unsigned lds_off = (unsigned)(uintptr_t)lds;   // low 32 bits of shared ptr = LDS offset
    asm volatile("global_load_async_to_lds_b128 %0, %1, off"
                 :: "v"(lds_off), "v"((unsigned long long)g) : "memory");
}
__device__ __forceinline__ void wait_async0() {
    asm volatile("s_wait_asynccnt 0x0" ::: "memory");
}

// Issue this thread's 2 async chunk copies for one A K-step tile (128 rows x 64B).
template<int MODE>
__device__ __forceinline__ void issue_atile_fill(_Float16* tile, int tid, int rowBase, int ks,
                                                 const _Float16* __restrict__ A_src,
                                                 const _Float16* __restrict__ zpage) {
    #pragma unroll
    for (int i = 0; i < 2; ++i) {
        int s  = tid + i * 256;            // 0..511 chunk id
        int r  = s >> 2;
        int ch = s & 3;
        int p  = rowBase + r;
        const _Float16* g;
        if (MODE == 0) {
            int t  = ks >> 3;              // tap 0..8
            int kb = ks & 7;               // cin block 0..7
            int dy = t / 3 - 1, dx = t % 3 - 1;
            int b = p >> 10, y = (p >> 5) & 31, x = p & 31;
            int yy = y + dy, xx = x + dx;
            bool ok = ((unsigned)yy < (unsigned)Hh) && ((unsigned)xx < (unsigned)Ww);
            g = ok ? (A_src + ((size_t)(((b << 10) | (yy << 5) | xx)) * Cc + kb * 32 + ch * 8))
                   : (zpage + ch * 8);     // zero page: border padding without predication
        } else {
            g = A_src + ((size_t)p * 512 + ks * 32 + ch * 8);
        }
        async_g2l_b128(g, &tile[r * 40 + ch * 8]);
    }
}

// B panel for one K-step: 16 fragment blocks = contiguous 16KB of packed Bp.
__device__ __forceinline__ void issue_btile_fill(_Float16* btile, int tid, int ks,
                                                 const _Float16* __restrict__ Bp) {
    const _Float16* src = Bp + (size_t)ks * 8192;   // 16 blocks * 512 halfs
    #pragma unroll
    for (int i = 0; i < 4; ++i) {
        int s = tid + i * 256;             // 0..1023 chunks of 16B
        async_g2l_b128(src + s * 8, &btile[s * 8]);
    }
}

// ---------------------------------------------------------------------------
// WMMA GEMM: Craw[M,256] = A @ B + bias, with fused BN-stat atomics.
//   MODE 0: conv implicit-GEMM.  A rows are pixels; K-step ks -> (tap, kblock).
//   MODE 1: graph GEMM.          A = feat [N,512] f16; K-step ks -> kblock.
// Workgroup: 256 threads = 8 waves; tile 128(M) x 256(N).
// Wave (waveM in 0..3, waveN in 0..1) owns 32(M) x 128(N): acc[2][8] v8f.
// Both A tile and B panel double-buffered in LDS, filled by async
// global->LDS copies issued one K-step ahead: all global latency hidden.
// ---------------------------------------------------------------------------
template<int MODE>
__global__ __launch_bounds__(256)
void k_wmma_gemm(const _Float16* __restrict__ A_src,
                 const _Float16* __restrict__ Bp,
                 const float*    __restrict__ bias,
                 float*          __restrict__ Craw,
                 float*          __restrict__ bnsum,
                 float*          __restrict__ bnsq,
                 const _Float16* __restrict__ zpage) {
    __shared__ __align__(16) _Float16 atile[2][128 * 40];  // 2 x 10 KB
    __shared__ __align__(16) _Float16 btile[2][16 * 512];  // 2 x 16 KB

    const int tid   = threadIdx.x;
    const int lane  = tid & 31;
    const int wave  = tid >> 5;
    const int waveM = wave & 3;
    const int waveN = wave >> 2;
    const int rowBase = blockIdx.x * 128;

    v8f acc[2][8] = {};

    const int laneRow = lane & 15;
    const int laneK0  = (lane & 16) ? 16 : 0;   // A-frag halfword group
    const int KSTEPS  = (MODE == 0) ? KCONV_STEPS : KGRAPH_STEPS;

    // prologue: start async fills of buffer 0
    issue_atile_fill<MODE>(atile[0], tid, rowBase, 0, A_src, zpage);
    issue_btile_fill(btile[0], tid, 0, Bp);

    for (int ks = 0; ks < KSTEPS; ++ks) {
        wait_async0();                 // this wave's fill(ks) landed in LDS
        __syncthreads();               // everyone's fill(ks) landed; prev compute done

        const _Float16* acur = atile[ks & 1];
        const _Float16* bcur = btile[ks & 1];
        if (ks + 1 < KSTEPS) {         // overlap next tile fetch with compute below
            issue_atile_fill<MODE>(atile[(ks + 1) & 1], tid, rowBase, ks + 1, A_src, zpage);
            issue_btile_fill(btile[(ks + 1) & 1], tid, ks + 1, Bp);
        }
        if (ks + 2 < KSTEPS) {         // warm L2 for the panel after next (thread-strided)
            __builtin_prefetch(Bp + (size_t)(ks + 2) * 8192 + tid * 32, 0, 2);
        }

        // ---- A fragments from LDS (two b128 per frag per lane) ----
        FragH a[2];
        #pragma unroll
        for (int mi = 0; mi < 2; ++mi) {
            int r = waveM * 32 + mi * 16 + laneRow;
            const _Float16* base = &acur[r * 40 + laneK0];
            a[mi].u[0] = *(const uint4*)(base);
            a[mi].u[1] = *(const uint4*)(base + 8);
        }

        // ---- B fragments from LDS + 16 WMMAs ----
        const _Float16* bbase = bcur + (size_t)(waveN * 8) * 512 + lane * 16;
        #pragma unroll
        for (int j = 0; j < 8; ++j) {
            FragH bf;
            const _Float16* bb = bbase + (size_t)j * 512;
            bf.u[0] = *(const uint4*)(bb);
            bf.u[1] = *(const uint4*)(bb + 8);
            #pragma unroll
            for (int mi = 0; mi < 2; ++mi) {
                acc[mi][j] = __builtin_amdgcn_wmma_f32_16x16x32_f16(
                    false, a[mi].h, false, bf.h, (short)0, acc[mi][j], false, false);
            }
        }
    }

    // ---- epilogue: +bias, store f32, fused BN sum/sumsq atomics ----
    const int rowAdd = (lane & 16) ? 8 : 0;
    #pragma unroll
    for (int j = 0; j < 8; ++j) {
        int col = waveN * 128 + j * 16 + (lane & 15);
        float bv = bias[col];
        float s = 0.0f, q = 0.0f;
        #pragma unroll
        for (int mi = 0; mi < 2; ++mi) {
            int r0 = rowBase + waveM * 32 + mi * 16 + rowAdd;
            #pragma unroll
            for (int v = 0; v < 8; ++v) {
                float val = acc[mi][j][v] + bv;
                Craw[(size_t)(r0 + v) * Cc + col] = val;
                s += val;
                q += val * val;
            }
        }
        atomicAdd(&bnsum[col], s);
        atomicAdd(&bnsq[col], q);
    }
}

// ---------------------------------------------------------------------------
// BN finalize: per-channel scale/shift from sums
// ---------------------------------------------------------------------------
__global__ void k_bn_finalize(const float* __restrict__ sum, const float* __restrict__ sq,
                              const float* __restrict__ gamma, const float* __restrict__ beta,
                              float* __restrict__ sc, float* __restrict__ sh, float invn) {
    int c = threadIdx.x;
    if (c >= Cc) return;
    float mean = sum[c] * invn;
    float var  = sq[c] * invn - mean * mean;
    float inv  = rsqrtf(var + 1e-5f);
    float s    = gamma[c] * inv;
    sc[c] = s;
    sh[c] = beta[c] - mean * s;
}

// BN2d + ReLU applied to conv raw -> nodes f16, written into feat[:,256:512]
__global__ void k_bn_relu_nodes(const float* __restrict__ craw,
                                const float* __restrict__ sc, const float* __restrict__ sh,
                                _Float16* __restrict__ feat) {
    int idx = blockIdx.x * blockDim.x + threadIdx.x;   // over N*64, 4 ch each
    if (idx >= Nn * (Cc / 4)) return;
    int node = idx / (Cc / 4);
    int c0   = (idx % (Cc / 4)) * 4;
    const float4 r = *(const float4*)&craw[(size_t)node * Cc + c0];
    H4 o;
    o.h[0] = (_Float16)fmaxf(r.x * sc[c0 + 0] + sh[c0 + 0], 0.0f);
    o.h[1] = (_Float16)fmaxf(r.y * sc[c0 + 1] + sh[c0 + 1], 0.0f);
    o.h[2] = (_Float16)fmaxf(r.z * sc[c0 + 2] + sh[c0 + 2], 0.0f);
    o.h[3] = (_Float16)fmaxf(r.w * sc[c0 + 3] + sh[c0 + 3], 0.0f);
    *(uint2*)&feat[(size_t)node * 512 + 256 + c0] = o.u;
}

// ---------------------------------------------------------------------------
// segment_sum: for each edge, agg[dst] += nodes[src] (nodes = feat[:,256:512])
// 32 threads per edge, 8 channels each (one b128 f16 gather per thread).
// ---------------------------------------------------------------------------
__global__ void k_scatter_add(const int* __restrict__ eidx, int E,
                              const _Float16* __restrict__ feat,
                              float* __restrict__ agg) {
    long long tid = (long long)blockIdx.x * blockDim.x + threadIdx.x;
    if (tid >= (long long)E * 32) return;
    int e    = (int)(tid >> 5);
    int c0   = (int)(tid & 31) * 8;
    int src  = eidx[e];
    int dst  = eidx[E + e];
    FragH v;  // only u[0] used (8 halfs)
    v.u[0] = *(const uint4*)&feat[(size_t)src * 512 + 256 + c0];
    float* out = &agg[(size_t)dst * Cc + c0];
    #pragma unroll
    for (int i = 0; i < 8; ++i)
        atomicAdd(&out[i], (float)v.h[i]);
}

// agg f32 -> feat[:,0:256] f16
__global__ void k_agg_to_f16(const float* __restrict__ agg, _Float16* __restrict__ feat) {
    int idx = blockIdx.x * blockDim.x + threadIdx.x;   // over N*64
    if (idx >= Nn * (Cc / 4)) return;
    int node = idx / (Cc / 4);
    int c0   = (idx % (Cc / 4)) * 4;
    const float4 r = *(const float4*)&agg[(size_t)node * Cc + c0];
    H4 o;
    o.h[0] = (_Float16)r.x; o.h[1] = (_Float16)r.y;
    o.h[2] = (_Float16)r.z; o.h[3] = (_Float16)r.w;
    *(uint2*)&feat[(size_t)node * 512 + c0] = o.u;
}

// ---------------------------------------------------------------------------
// Final: out NCHW = relu(graw * sc1 + sh1) + x   (coalesced over NCHW index)
// ---------------------------------------------------------------------------
__global__ void k_final(const float* __restrict__ graw,
                        const float* __restrict__ sc, const float* __restrict__ sh,
                        const float* __restrict__ x, float* __restrict__ out) {
    int idx = blockIdx.x * blockDim.x + threadIdx.x;   // over B*C*H*W
    if (idx >= Bb * Cc * HW) return;
    int hw = idx & (HW - 1);
    int c  = (idx >> 10) & (Cc - 1);
    int b  = idx >> 18;
    int node = (b << 10) | hw;
    float g = graw[(size_t)node * Cc + c];
    out[idx] = fmaxf(g * sc[c] + sh[c], 0.0f) + x[idx];
}

// ---------------------------------------------------------------------------
// Host-side launch
// ---------------------------------------------------------------------------
extern "C" void kernel_launch(void* const* d_in, const int* in_sizes, int n_in,
                              void* d_out, int out_size, void* d_ws, size_t ws_size,
                              hipStream_t stream) {
    const float* x       = (const float*)d_in[0];
    const int*   eidx    = (const int*)  d_in[1];
    const float* conv_w  = (const float*)d_in[2];
    const float* conv_b  = (const float*)d_in[3];
    const float* bn2d_g  = (const float*)d_in[4];
    const float* bn2d_b  = (const float*)d_in[5];
    const float* w_rel   = (const float*)d_in[6];
    const float* b_rel   = (const float*)d_in[7];
    const float* w_root  = (const float*)d_in[8];
    const float* bn1d_g  = (const float*)d_in[9];
    const float* bn1d_b  = (const float*)d_in[10];
    float* out = (float*)d_out;
    const int E = in_sizes[1] / 2;

    char* ws = (char*)d_ws;
    _Float16* xh    = (_Float16*)(ws + O_XH);
    float*    craw  = (float*)   (ws + O_CRAW);    // conv raw, later graph raw
    _Float16* feat  = (_Float16*)(ws + O_FEAT);
    float*    agg   = (float*)   (ws + O_AGG);
    _Float16* wconv = (_Float16*)(ws + O_WCONV);
    _Float16* wg    = (_Float16*)(ws + O_WG);
    float*    st    = (float*)   (ws + O_STATS);
    const _Float16* zpage = (const _Float16*)(st + S_ZPAGE);

    const int T = 256;
    // zero accumulators + stats + zero page
    k_zero<<<2048, T, 0, stream>>>(agg, Nn * Cc);
    k_zero<<<9,    T, 0, stream>>>(st, 2048 + 64);

    // weight packing + input conversion
    k_pack_conv_w <<<(9 * Cc * Cc + T - 1) / T, T, 0, stream>>>(conv_w, wconv);
    k_pack_graph_w<<<(512 * Cc + T - 1) / T,    T, 0, stream>>>(w_rel, w_root, wg);
    k_x_to_nhwc   <<<(Bb * Cc * HW) / T,        T, 0, stream>>>(x, xh);

    // conv as implicit WMMA GEMM (+ fused BN2d stats)
    k_wmma_gemm<0><<<Nn / 128, T, 0, stream>>>(xh, wconv, conv_b, craw,
                                               st + S_SUM2D, st + S_SQ2D, zpage);
    k_bn_finalize<<<1, T, 0, stream>>>(st + S_SUM2D, st + S_SQ2D, bn2d_g, bn2d_b,
                                       st + S_SC2D, st + S_SH2D, 1.0f / (float)Nn);
    k_bn_relu_nodes<<<(Nn * (Cc / 4)) / T, T, 0, stream>>>(craw, st + S_SC2D, st + S_SH2D, feat);

    // graph aggregation (gather + float atomics), then f16 pack of agg
    k_scatter_add<<<(int)(((long long)E * 32 + T - 1) / T), T, 0, stream>>>(eidx, E, feat, agg);
    k_agg_to_f16 <<<(Nn * (Cc / 4)) / T, T, 0, stream>>>(agg, feat);

    // fused GraphConv GEMM: [agg|nodes] @ [Wrel|Wroot]^T + b_rel (+ BN1d stats)
    k_wmma_gemm<1><<<Nn / 128, T, 0, stream>>>(feat, wg, b_rel, craw,
                                               st + S_SUM1D, st + S_SQ1D, zpage);
    k_bn_finalize<<<1, T, 0, stream>>>(st + S_SUM1D, st + S_SQ1D, bn1d_g, bn1d_b,
                                       st + S_SC1D, st + S_SH1D, 1.0f / (float)Nn);

    // BN1d + ReLU + residual back to NCHW
    k_final<<<(Bb * Cc * HW) / T, T, 0, stream>>>(craw, st + S_SC1D, st + S_SH1D, x, out);
}
